// MultiHeadAttention_17712445129205
// MI455X (gfx1250) — matrix operand order repaired and checked by reading, not compiled, not run on hardware
//
#include <hip/hip_runtime.h>

// ---------------------------------------------------------------------------
// MultiHeadAttention on MI455X (gfx1250): bf16 WMMA flash attention
//   B=8, T=2048, C=400 (H=4 heads x D=100), fp32 in/out
// Padded dims: Cpad=416 (13 x K32), Dpad=128 (4 x K32, 8 N-tiles)
// Round 3: flash processes 32 keys/iteration so the PV WMMA runs with a
// full K=32 P-matrix (round 2 zero-padded 16 keys -> 50% WMMA waste).
// ---------------------------------------------------------------------------

typedef __attribute__((ext_vector_type(16))) __bf16 v16bf;
typedef __attribute__((ext_vector_type(8)))  __bf16 v8bf;
typedef __attribute__((ext_vector_type(8)))  float  v8f;

// explicit global-address-space pointer types -> global_load_*/global_store_*
typedef const __bf16 __attribute__((address_space(1)))* gcbf;
typedef __bf16       __attribute__((address_space(1)))* gbf;
typedef const float  __attribute__((address_space(1)))* gcf;
typedef float        __attribute__((address_space(1)))* gf;
typedef const v16bf  __attribute__((address_space(1)))* gcv16;
typedef const v8bf   __attribute__((address_space(1)))* gcv8;

#define TT   2048
#define CC   400
#define CP   416
#define DD   100
#define DP   128
#define HH   4
#define BB   8
#define BT   (BB*TT)          // 16384

// workspace layout (all sizes multiples of 256B)
#define SZ_XBF  ((size_t)BT*CP*2)        // 13,631,488
#define SZ_WT   ((size_t)HH*DP*CP*2)     //    425,984 per W
#define SZ_WO   ((size_t)CC*CP*2)        //    332,800
#define SZ_QKV  ((size_t)BB*HH*TT*DP*2)  // 16,777,216 each
#define OFF_XBF  ((size_t)0)
#define OFF_WQ   (OFF_XBF + SZ_XBF)
#define OFF_WK   (OFF_WQ  + SZ_WT)
#define OFF_WV   (OFF_WK  + SZ_WT)
#define OFF_WO   (OFF_WV  + SZ_WT)
#define OFF_Q    (OFF_WO  + SZ_WO)
#define OFF_K    (OFF_Q   + SZ_QKV)
#define OFF_VT   (OFF_K   + SZ_QKV)
#define OFF_ATT  (OFF_VT  + SZ_QKV)

static __device__ inline v8f wmma_bf16(v16bf a, v16bf b, v8f c) {
  // D = A(16x32 bf16) x B(32x16 bf16) + C(16x16 f32)
  return __builtin_amdgcn_wmma_f32_16x16x32_bf16(
      /*neg_a=*/false, a, /*neg_b=*/false, b,
      /*c_mod=*/(short)0, c, /*reuse_a=*/false, /*reuse_b=*/false);
}

// A-fragment (16-bit, 16x32): lane L holds row M=L%16,
// K in [kbase+16*half, +16) contiguous (elems 0..15)
static __device__ inline v16bf load_a_frag(gcbf row, int kbase, int half) {
  v8bf lo = *(gcv8)(row + kbase + 16 * half);
  v8bf hi = *(gcv8)(row + kbase + 16 * half + 8);
  v16bf r;
#pragma unroll
  for (int i = 0; i < 8; ++i) { r[i] = lo[i]; r[i + 8] = hi[i]; }
  return r;
}

static __device__ inline v8f zero8() {
  v8f z = {0.f, 0.f, 0.f, 0.f, 0.f, 0.f, 0.f, 0.f};
  return z;
}

// ---------------------------------------------------------------------------
// Kernel 0: fp32 -> bf16 staging with zero padding
// ---------------------------------------------------------------------------
__global__ void mha_prep_kernel(const float* __restrict__ x,
                                const float* __restrict__ Wq,
                                const float* __restrict__ Wk,
                                const float* __restrict__ Wv,
                                const float* __restrict__ Wo,
                                __bf16* __restrict__ xbf,
                                __bf16* __restrict__ wqt,
                                __bf16* __restrict__ wkt,
                                __bf16* __restrict__ wvt,
                                __bf16* __restrict__ wobf,
                                __bf16* __restrict__ attnbf) {
  const long SEG0 = (long)BT * CP;
  const long SEG1 = (long)HH * DP * CP;
  const long SEG2 = (long)CC * CP;
  const long SEG3 = (long)BT * 16;
  const long NTOT = SEG0 + 3 * SEG1 + SEG2 + SEG3;
  for (long i = (long)blockIdx.x * blockDim.x + threadIdx.x; i < NTOT;
       i += (long)gridDim.x * blockDim.x) {
    long idx = i;
    if (idx < SEG0) {
      long row = idx / CP; int c = (int)(idx % CP);
      ((gbf)xbf)[idx] = (c < CC) ? (__bf16)((gcf)x)[row * CC + c] : (__bf16)0.0f;
      continue;
    }
    idx -= SEG0;
    if (idx < 3 * SEG1) {
      int w = (int)(idx / SEG1); long r = idx % SEG1;
      int h = (int)(r / ((long)DP * CP)); int rem = (int)(r % ((long)DP * CP));
      int d = rem / CP, c = rem % CP;
      const float* W = (w == 0) ? Wq : (w == 1) ? Wk : Wv;
      __bf16* Wt = (w == 0) ? wqt : (w == 1) ? wkt : wvt;
      float v = (d < DD && c < CC) ? ((gcf)W)[((long)h * CC + c) * DD + d] : 0.0f;
      ((gbf)Wt)[((long)h * DP + d) * CP + c] = (__bf16)v;
      continue;
    }
    idx -= 3 * SEG1;
    if (idx < SEG2) {
      long n = idx / CP; int c = (int)(idx % CP);
      ((gbf)wobf)[idx] = (c < CC) ? (__bf16)((gcf)Wo)[n * CC + c] : (__bf16)0.0f;
      continue;
    }
    idx -= SEG2;
    {
      long row = idx / 16; int c = CC + (int)(idx % 16);
      ((gbf)attnbf)[row * CP + c] = (__bf16)0.0f;
    }
  }
}

// ---------------------------------------------------------------------------
// Kernel 1: QKV projection GEMM. One wave per (b,h,16-row tile).
//   Q,K: [bh][T][DP] row-major bf16; V stored transposed: Vt [bh][DP][T]
// MODE 0/1: row-major [T][DP] store; MODE 2: transposed [DP][T] store.
// ---------------------------------------------------------------------------
template <int MODE>
static __device__ inline void proj_pass(gcbf arow, gcbf W, gbf dst,
                                        int m0, int lr, int half) {
  v8f acc[8];
#pragma unroll
  for (int j = 0; j < 8; ++j) acc[j] = zero8();

#pragma unroll 1                      // bound A-frag live range (no spills)
  for (int ks = 0; ks < 13; ++ks) {   // K = 416 = 13 x 32
    const int cb = ks * 32;
    v16bf a = load_a_frag(arow, cb, half);
#pragma unroll
    for (int j = 0; j < 8; ++j) {     // N = 128 = 8 x 16
      // B-fragment: lane -> N=lr (output col d), 16 contiguous K values
      v16bf bf = *(gcv16)(W + (size_t)(j * 16 + lr) * CP + cb + 16 * half);
      acc[j] = wmma_bf16(a, bf, acc[j]);
    }
  }
  // C layout: lane -> N=lr, reg r -> M=r+8*half
#pragma unroll
  for (int j = 0; j < 8; ++j) {
#pragma unroll
    for (int r = 0; r < 8; ++r) {
      const int m = m0 + r + 8 * half;
      const int d = j * 16 + lr;
      const __bf16 v = (__bf16)acc[j][r];
      if (MODE == 2) dst[(size_t)d * TT + m] = v;
      else           dst[(size_t)m * DP + d] = v;
    }
  }
}

__global__ void mha_qkv_proj_kernel(const __bf16* __restrict__ xbf,
                                    const __bf16* __restrict__ wqt,
                                    const __bf16* __restrict__ wkt,
                                    const __bf16* __restrict__ wvt,
                                    __bf16* __restrict__ Q,
                                    __bf16* __restrict__ K,
                                    __bf16* __restrict__ Vt) {
  const int lane = threadIdx.x;
  const int half = lane >> 4, lr = lane & 15;
  const int wave = blockIdx.x;           // 4096 waves
  const int bh = wave >> 7;              // /128
  const int mt = wave & 127;
  const int b = bh >> 2, h = bh & 3;
  const int m0 = mt * 16;

  gcbf arow = (gcbf)xbf + (size_t)(b * TT + m0 + lr) * CP;
  const size_t wofs = (size_t)h * DP * CP;
  const size_t qofs = (size_t)bh * TT * DP;

  proj_pass<0>(arow, (gcbf)wqt + wofs, (gbf)Q + qofs, m0, lr, half);
  proj_pass<1>(arow, (gcbf)wkt + wofs, (gbf)K + qofs, m0, lr, half);
  proj_pass<2>(arow, (gcbf)wvt + wofs, (gbf)Vt + (size_t)bh * DP * TT,
               m0, lr, half);
}

// ---------------------------------------------------------------------------
// Kernel 2: causal flash attention. One wave per (b,h,16-query tile).
// 32 keys per iteration: S = Q Kblk^T as two 16-wide tiles (8 WMMA), online
// softmax with folded reductions, P(16x32) via LDS transpose, O += P Vblk
// (8 WMMA, full K=32 utilization).
// ---------------------------------------------------------------------------
__global__ void mha_flash_attn_kernel(const __bf16* __restrict__ Q,
                                      const __bf16* __restrict__ K,
                                      const __bf16* __restrict__ Vt,
                                      __bf16* __restrict__ attnbf) {
  __shared__ __align__(64) __bf16 plds[16][32];   // P tile: 16 rows x 32 keys

  const int lane = threadIdx.x;
  const int half = lane >> 4, lr = lane & 15;
  const int wave = blockIdx.x;           // 4096 waves
  const int bh = wave >> 7;
  const int mt = wave & 127;
  const int b = bh >> 2, h = bh & 3;
  const int m0 = mt * 16;

  gcbf Qb = (gcbf)Q  + (size_t)bh * TT * DP;
  gcbf Kb = (gcbf)K  + (size_t)bh * TT * DP;
  gcbf Vb = (gcbf)Vt + (size_t)bh * DP * TT;
  gbf  Ab = (gbf)attnbf;

  // Q A-fragments for all 4 K-steps (d = 0..127), kept resident
  v16bf aq[4];
#pragma unroll
  for (int ks = 0; ks < 4; ++ks)
    aq[ks] = load_a_frag(Qb + (size_t)(m0 + lr) * DP, ks * 32, half);

  v8f O[8];
#pragma unroll
  for (int j = 0; j < 8; ++j) O[j] = zero8();
  float mrun[8], lrun[8];
#pragma unroll
  for (int r = 0; r < 8; ++r) { mrun[r] = -__builtin_inff(); lrun[r] = 0.0f; }

  const float scale = 0.1f;   // 1/sqrt(100)
  const int nblocks = (mt >> 1) + 1;   // 32-key blocks covering keys <= m0+15

#pragma unroll 1
  for (int sb = 0; sb < nblocks; ++sb) {
    const int s0 = sb * 32;
    if (sb + 1 < nblocks)  // prefetch next key block
      __builtin_prefetch(Kb + (size_t)(s0 + 32 + lr) * DP, 0, 1);

    // S block = Q(16x128) x K^T, two 16-key tiles sharing resident aq[]
    v8f slo = zero8(), shi = zero8();
#pragma unroll
    for (int ks = 0; ks < 4; ++ks) {
      v16bf bk = *(gcv16)(Kb + (size_t)(s0 + lr) * DP + ks * 32 + 16 * half);
      slo = wmma_bf16(aq[ks], bk, slo);
    }
#pragma unroll
    for (int ks = 0; ks < 4; ++ks) {
      v16bf bk = *(gcv16)(Kb + (size_t)(s0 + 16 + lr) * DP + ks * 32 + 16 * half);
      shi = wmma_bf16(aq[ks], bk, shi);
    }

    const bool last = (sb == nblocks - 1);
    float plo[8], phi[8];
#pragma unroll
    for (int r = 0; r < 8; ++r) {
      const int mrow = m0 + r + 8 * half;
      float svl = slo[r] * scale;
      float svh = shi[r] * scale;
      if (last) {
        if (s0 + lr > mrow)      svl = -__builtin_inff();
        if (s0 + 16 + lr > mrow) svh = -__builtin_inff();
      }
      // folded row-max over 32 keys (16 lanes hold this row's N values)
      float rm = fmaxf(svl, svh);
      rm = fmaxf(rm, __shfl_xor(rm, 1, 32));
      rm = fmaxf(rm, __shfl_xor(rm, 2, 32));
      rm = fmaxf(rm, __shfl_xor(rm, 4, 32));
      rm = fmaxf(rm, __shfl_xor(rm, 8, 32));
      const float mnew = fmaxf(mrun[r], rm);
      const float pl = __expf(svl - mnew);
      const float ph = __expf(svh - mnew);
      float rs = pl + ph;               // folded row-sum
      rs += __shfl_xor(rs, 1, 32);
      rs += __shfl_xor(rs, 2, 32);
      rs += __shfl_xor(rs, 4, 32);
      rs += __shfl_xor(rs, 8, 32);
      const float alpha = __expf(mrun[r] - mnew);
      lrun[r] = lrun[r] * alpha + rs;
      mrun[r] = mnew;
#pragma unroll
      for (int j = 0; j < 8; ++j) O[j][r] *= alpha;
      plo[r] = pl;
      phi[r] = ph;
    }

    // C-layout -> A-layout transpose through LDS (DS ops are in-order per
    // wave; compiler barriers keep issue order, s_wait_dscnt covers RAW)
    asm volatile("" ::: "memory");
#pragma unroll
    for (int r = 0; r < 8; ++r) {
      plds[r + 8 * half][lr]      = (__bf16)plo[r];  // keys s0..s0+15
      plds[r + 8 * half][16 + lr] = (__bf16)phi[r];  // keys s0+16..s0+31
    }
    asm volatile("s_wait_dscnt 0" ::: "memory");

    v16bf pa;                                        // A-frag of P (16x32)
#pragma unroll
    for (int i = 0; i < 8; ++i) {
      pa[i]     = plds[lr][16 * half + i];
      pa[i + 8] = plds[lr][16 * half + 8 + i];
    }
    asm volatile("" ::: "memory");

    // O += P(16x32) x V(32x16): full-K PV, one clean 32B load per lane
#pragma unroll
    for (int j = 0; j < 8; ++j) {
      v16bf vb = *(gcv16)(Vb + (size_t)(j * 16 + lr) * TT + s0 + 16 * half);
      O[j] = wmma_bf16(pa, vb, O[j]);
    }
  }

  // normalize + write heads concatenated: attnbf[b*T+m][h*100+d], d<100
#pragma unroll
  for (int j = 0; j < 8; ++j) {
#pragma unroll
    for (int r = 0; r < 8; ++r) {
      const int d = j * 16 + lr;
      if (d < DD) {
        const int m = m0 + r + 8 * half;
        const float o = O[j][r] / lrun[r];
        Ab[((size_t)b * TT + m) * CP + h * DD + d] = (__bf16)o;
      }
    }
  }
}

// ---------------------------------------------------------------------------
// Kernel 3: output projection  out = attn x Wo^T + bo   (M=16384,N=400,K=416)
// One wave per (16-row tile, 128-col group); fp32 output.
// ---------------------------------------------------------------------------
__global__ void mha_out_proj_kernel(const __bf16* __restrict__ attnbf,
                                    const __bf16* __restrict__ wobf,
                                    const float* __restrict__ bo,
                                    float* __restrict__ out) {
  const int lane = threadIdx.x;
  const int half = lane >> 4, lr = lane & 15;
  const int wave = blockIdx.x;          // 4096 waves
  const int mt = wave & 1023;
  const int ng = wave >> 10;            // n-group: cols [ng*128, ...)
  const int m0 = mt * 16;
  const int nbase = ng * 128;

  v8f acc[8];
#pragma unroll
  for (int j = 0; j < 8; ++j) acc[j] = zero8();

  gcbf arow = (gcbf)attnbf + (size_t)(m0 + lr) * CP;
  gcbf Wb   = (gcbf)wobf;

#pragma unroll 1
  for (int ks = 0; ks < 13; ++ks) {
    const int cb = ks * 32;
    v16bf a = load_a_frag(arow, cb, half);
#pragma unroll
    for (int j = 0; j < 8; ++j) {
      if (nbase + j * 16 < CC) {
        // B[k=c][n] = Wo[n][c]: lane -> row n = nbase+j*16+lr, contiguous c
        v16bf bf = *(gcv16)(Wb + (size_t)(nbase + j * 16 + lr) * CP +
                            cb + 16 * half);
        acc[j] = wmma_bf16(a, bf, acc[j]);
      }
    }
  }

#pragma unroll
  for (int j = 0; j < 8; ++j) {
    if (nbase + j * 16 < CC) {
      const int n = nbase + j * 16 + lr;
      const float bias = ((gcf)bo)[n];
#pragma unroll
      for (int r = 0; r < 8; ++r) {
        const int m = m0 + r + 8 * half;
        ((gf)out)[(size_t)m * CC + n] = acc[j][r] + bias;
      }
    }
  }
}

// ---------------------------------------------------------------------------
extern "C" void kernel_launch(void* const* d_in, const int* in_sizes, int n_in,
                              void* d_out, int out_size, void* d_ws, size_t ws_size,
                              hipStream_t stream) {
  const float* x  = (const float*)d_in[0];
  const float* Wq = (const float*)d_in[1];
  const float* Wk = (const float*)d_in[2];
  const float* Wv = (const float*)d_in[3];
  const float* Wo = (const float*)d_in[4];
  const float* bo = (const float*)d_in[5];
  float* out = (float*)d_out;

  char* ws = (char*)d_ws;
  __bf16* xbf    = (__bf16*)(ws + OFF_XBF);
  __bf16* wqt    = (__bf16*)(ws + OFF_WQ);
  __bf16* wkt    = (__bf16*)(ws + OFF_WK);
  __bf16* wvt    = (__bf16*)(ws + OFF_WV);
  __bf16* wobf   = (__bf16*)(ws + OFF_WO);
  __bf16* Qw     = (__bf16*)(ws + OFF_Q);
  __bf16* Kw     = (__bf16*)(ws + OFF_K);
  __bf16* Vtw    = (__bf16*)(ws + OFF_VT);
  __bf16* attnbf = (__bf16*)(ws + OFF_ATT);

  mha_prep_kernel<<<2048, 256, 0, stream>>>(x, Wq, Wk, Wv, Wo,
                                            xbf, wqt, wkt, wvt, wobf, attnbf);
  mha_qkv_proj_kernel<<<4096, 32, 0, stream>>>(xbf, wqt, wkt, wvt, Qw, Kw, Vtw);
  mha_flash_attn_kernel<<<4096, 32, 0, stream>>>(Qw, Kw, Vtw, attnbf);
  mha_out_proj_kernel<<<4096, 32, 0, stream>>>(attnbf, wobf, bo, out);
}
